// muPAttention_28870770163998
// MI455X (gfx1250) — matrix operand order repaired and checked by reading, not compiled, not run on hardware
//
#include <hip/hip_runtime.h>
#include <hip/hip_bf16.h>
#include <math.h>

// ---------------------------------------------------------------------------
// muP attention, B=2 S=2048 D=1024 H=16 Dk=64, bf16 WMMA pipeline on gfx1250.
// K/V tiles staged to LDS via the Tensor Data Mover (tensor_load_to_lds),
// double-buffered, TENSORcnt-tracked.
// ---------------------------------------------------------------------------

#define BB   2
#define SS   2048
#define DD   1024
#define HH   16
#define DK   64
#define MM   (BB * SS)          // 4096 rows for the projection GEMMs

typedef __attribute__((ext_vector_type(16))) __bf16 v16bf;
typedef __attribute__((ext_vector_type(8)))  float  v8f;
typedef __attribute__((ext_vector_type(4)))  unsigned int u32x4;
typedef __attribute__((ext_vector_type(8)))  int i32x8;
typedef __attribute__((ext_vector_type(4)))  int i32x4;

union Frag {
  v16bf v;
  uint4 u[2];
};

__device__ __forceinline__ v8f wmma_bf16(const Frag& a, const Frag& b, v8f c) {
  // (neg_a, A, neg_b, B, c_mod, C, reuse_a, reuse_b)
  return __builtin_amdgcn_wmma_f32_16x16x32_bf16(false, a.v, false, b.v,
                                                 (short)0, c, false, false);
}

// A-matrix (16x32 bf16, MxK): lanes 0-15 -> rows, chunks K=[8h,8h+8) and [16+8h,...)
__device__ __forceinline__ void load_a_frag(Frag& f, const __bf16* __restrict__ base,
                                            int row, int ld, int k0, int halfSel) {
  const __bf16* p = base + (size_t)row * ld + k0 + halfSel * 8;
  f.u[0] = *reinterpret_cast<const uint4*>(p);
  f.u[1] = *reinterpret_cast<const uint4*>(p + 16);
}

// B-matrix (32x16 bf16, KxN) loaded as rows of B^T: lane n holds K=[8h,8h+8) and [16+8h,...)
__device__ __forceinline__ void load_b_frag(Frag& f, const __bf16* __restrict__ base,
                                            int row, int ld, int k0, int halfSel) {
  const __bf16* p = base + (size_t)row * ld + k0 + halfSel * 8;
  f.u[0] = *reinterpret_cast<const uint4*>(p);
  f.u[1] = *reinterpret_cast<const uint4*>(p + 16);
}

// Same B-fragment pattern but sourced from an LDS tile with row stride 64 bf16.
__device__ __forceinline__ void load_b_frag_lds(Frag& f, const __bf16* base,
                                                int row, int k0, int halfSel) {
  const __bf16* p = base + row * 64 + k0 + halfSel * 8;
  f.u[0] = *reinterpret_cast<const uint4*>(p);
  f.u[1] = *reinterpret_cast<const uint4*>(p + 16);
}

// ---------------------------------------------------------------------------
// Tensor Data Mover: 2-D tile load, bf16 elements (data_size=2B).
// D# group0: count=1, lds_addr, 57-bit global_addr, type=2.
// D# group1: data_size=1, tensor_dim0/1, tile_dim0/1, tensor_dim0_stride.
// Groups 2/3 (and the trailing group operand) zero: 2-D tensor.
// This toolchain exposes the 6-argument builtin:
//   (u32x4 g0, i32x8 g1, i32x4 g2, i32x4 g3, i32x8 g4, i32 cpol)
// ---------------------------------------------------------------------------
__device__ __forceinline__ void tdm_load_2d(unsigned ldsAddr, const void* gptr,
                                            unsigned dim0, unsigned dim1,
                                            unsigned long long stride0,
                                            unsigned tile0, unsigned tile1) {
  const unsigned long long ga = (unsigned long long)(uintptr_t)gptr;
  u32x4 g0;
  g0[0] = 1u;                                             // count=1, user mode
  g0[1] = ldsAddr;                                        // LDS byte address
  g0[2] = (unsigned)(ga & 0xFFFFFFFFu);                   // global_addr[31:0]
  g0[3] = (unsigned)((ga >> 32) & 0x01FFFFFFu) | 0x80000000u;  // [56:32] | type=2
  i32x8 g1;
  g1[0] = 0x00010000;                                     // data_size=1 (2 bytes)
  g1[1] = (int)((dim0 & 0xFFFFu) << 16);                  // tensor_dim0[15:0] @ bit48
  g1[2] = (int)((dim0 >> 16) | ((dim1 & 0xFFFFu) << 16)); // dim0 hi | dim1 lo
  g1[3] = (int)((dim1 >> 16) | (tile0 << 16));            // dim1 hi | tile_dim0
  g1[4] = (int)tile1;                                     // tile_dim1 (tile_dim2=0)
  g1[5] = (int)(unsigned)(stride0 & 0xFFFFFFFFu);         // dim0_stride lo
  g1[6] = (int)(unsigned)((stride0 >> 32) & 0xFFFFu);     // dim0_stride hi
  g1[7] = 0;
  i32x4 z4 = {0, 0, 0, 0};
  i32x8 z8 = {0, 0, 0, 0, 0, 0, 0, 0};
  __builtin_amdgcn_tensor_load_to_lds(g0, g1, z4, z4, z8, 0);
}

// ---------------------------------------------------------------------------
// f32 -> bf16 conversion
// ---------------------------------------------------------------------------
__global__ __launch_bounds__(256) void cvt_f32_bf16(const float* __restrict__ in,
                                                    __bf16* __restrict__ out, int n) {
  int i = blockIdx.x * 256 + threadIdx.x;
  if (i < n) out[i] = (__bf16)in[i];
}

// ---------------------------------------------------------------------------
// Y = X[M,K] * W[N,K]^T ; bf16 inputs, f32 accumulate via v_wmma_f32_16x16x32_bf16
// mode 0: bf16 out, Q/K layout  [(b*H+h), s, dk]
// mode 1: bf16 out, V^T layout  [(b*H+h), dk, s]
// mode 2: f32  out, plain [M,N]
// Block = 128 threads (4 waves, 2x2 of 32x32) -> 64x64 tile per block.
// ---------------------------------------------------------------------------
__global__ __launch_bounds__(128) void gemm_bf16(const __bf16* __restrict__ X,
                                                 const __bf16* __restrict__ W,
                                                 void* __restrict__ Y,
                                                 int M, int N, int K, int mode) {
  const int lane = threadIdx.x & 31;
  const int wave = threadIdx.x >> 5;
  const int halfSel = lane >> 4;
  const int rIT = lane & 15;

  const int mWave = blockIdx.x * 64 + (wave >> 1) * 32;
  const int nWave = blockIdx.y * 64 + (wave & 1) * 32;

  v8f acc[2][2];
#pragma unroll
  for (int i = 0; i < 2; ++i)
#pragma unroll
    for (int j = 0; j < 2; ++j) acc[i][j] = {};

  for (int k0 = 0; k0 < K; k0 += 32) {
    Frag a[2], b[2];
#pragma unroll
    for (int i = 0; i < 2; ++i)
      load_a_frag(a[i], X, mWave + i * 16 + rIT, K, k0, halfSel);
#pragma unroll
    for (int j = 0; j < 2; ++j)
      load_b_frag(b[j], W, nWave + j * 16 + rIT, K, k0, halfSel);
#pragma unroll
    for (int i = 0; i < 2; ++i)
#pragma unroll
      for (int j = 0; j < 2; ++j) acc[i][j] = wmma_bf16(a[i], b[j], acc[i][j]);
  }

#pragma unroll
  for (int i = 0; i < 2; ++i)
#pragma unroll
    for (int j = 0; j < 2; ++j)
#pragma unroll
      for (int v = 0; v < 8; ++v) {
        const int m = mWave + i * 16 + halfSel * 8 + v;
        const int n = nWave + j * 16 + rIT;
        const float val = acc[i][j][v];
        if (mode == 2) {
          reinterpret_cast<float*>(Y)[(size_t)m * N + n] = val;
        } else {
          const __bf16 bv = (__bf16)val;
          const int bIdx = m >> 11, s = m & (SS - 1);
          const int h = n >> 6, dk = n & (DK - 1);
          if (mode == 0)
            reinterpret_cast<__bf16*>(Y)[(((size_t)(bIdx * HH + h)) * SS + s) * DK + dk] = bv;
          else
            reinterpret_cast<__bf16*>(Y)[(((size_t)(bIdx * HH + h)) * DK + dk) * SS + s] = bv;
        }
      }
}

// ---------------------------------------------------------------------------
// Flash attention core. Grid: (B*H, S/64). Block: 128 threads (4 waves).
// Each wave: 16 query rows x Dk=64. Q,K: [BH, S, 64] bf16. Vt: [BH, 64, S] bf16.
// K/V 64x64 tiles are staged to LDS by the TDM (wave 0), double-buffered, so
// the 4 waves share one fetch instead of 4 redundant vector-load streams.
// Output: bf16 [(b,s), (h,dk)] = [4096, 1024] for the output projection.
// ---------------------------------------------------------------------------
__global__ __launch_bounds__(128) void flash_attn(const __bf16* __restrict__ Q,
                                                  const __bf16* __restrict__ Kk,
                                                  const __bf16* __restrict__ Vt,
                                                  __bf16* __restrict__ O) {
  const int bh = blockIdx.x;
  const int qTile = blockIdx.y;
  const int lane = threadIdx.x & 31;
  const int wave = threadIdx.x >> 5;
  const int halfSel = lane >> 4;
  const int rIT = lane & 15;
  const int qRowBase = qTile * 64 + wave * 16;

  const __bf16* __restrict__ Qb = Q + (size_t)bh * SS * DK;
  const __bf16* __restrict__ Kb = Kk + (size_t)bh * SS * DK;
  const __bf16* __restrict__ Vb = Vt + (size_t)bh * DK * SS;

  __shared__ __bf16 kTile[2][64][64];  // [key][d]  8 KB x2
  __shared__ __bf16 vTile[2][64][64];  // [d][key]  8 KB x2
  __shared__ __bf16 pLDS[4][16][64];   // per-wave P staging, 8 KB

  Frag qa[2];
#pragma unroll
  for (int c = 0; c < 2; ++c)
    load_a_frag(qa[c], Qb, qRowBase + rIT, DK, c * 32, halfSel);

  v8f accO[4];
#pragma unroll
  for (int t = 0; t < 4; ++t) accO[t] = {};
  float rowMax[8], rowSum[8];
#pragma unroll
  for (int v = 0; v < 8; ++v) { rowMax[v] = -INFINITY; rowSum[v] = 0.0f; }

  const float scale = 0.125f;  // 1/sqrt(64)

  // Prologue: TDM-stage the first K/V tiles into buffer 0.
  if (wave == 0) {
    tdm_load_2d((unsigned)(uintptr_t)&kTile[0][0][0], Kb, DK, SS, DK, 64, 64);
    tdm_load_2d((unsigned)(uintptr_t)&vTile[0][0][0], Vb, SS, DK, SS, 64, 64);
  }

  for (int kv = 0; kv < SS; kv += 64) {
    const int buf = (kv >> 6) & 1;

    if (wave == 0) {
      if (kv + 64 < SS) {
        // Prefetch next tile pair into the other buffer, then wait for current.
        tdm_load_2d((unsigned)(uintptr_t)&kTile[buf ^ 1][0][0],
                    Kb + (size_t)(kv + 64) * DK, DK, SS, DK, 64, 64);
        tdm_load_2d((unsigned)(uintptr_t)&vTile[buf ^ 1][0][0],
                    Vb + (size_t)(kv + 64), SS, DK, SS, 64, 64);
        __builtin_amdgcn_s_wait_tensorcnt(2);  // current pair complete (in-order)
      } else {
        __builtin_amdgcn_s_wait_tensorcnt(0);
      }
    }
    __syncthreads();  // K/V tiles visible to all waves

    // ---- scores S = Q * K^T for 64 keys (4 x 16-key subtiles), K from LDS ----
    v8f sAcc[4];
#pragma unroll
    for (int nt = 0; nt < 4; ++nt) {
      Frag kb0, kb1;
      load_b_frag_lds(kb0, &kTile[buf][0][0], nt * 16 + rIT, 0, halfSel);
      load_b_frag_lds(kb1, &kTile[buf][0][0], nt * 16 + rIT, 32, halfSel);
      v8f z = {};
      z = wmma_bf16(qa[0], kb0, z);
      z = wmma_bf16(qa[1], kb1, z);
      sAcc[nt] = z;
    }

    // ---- online softmax: row max across 4 subtiles + 16 lanes of the group ----
    float newMax[8];
#pragma unroll
    for (int v = 0; v < 8; ++v) {
      float mx = rowMax[v];
#pragma unroll
      for (int nt = 0; nt < 4; ++nt) mx = fmaxf(mx, sAcc[nt][v] * scale);
#pragma unroll
      for (int off = 1; off < 16; off <<= 1)
        mx = fmaxf(mx, __shfl_xor(mx, off, 32));
      newMax[v] = mx;
    }
#pragma unroll
    for (int v = 0; v < 8; ++v) {
      const float corr = __expf(rowMax[v] - newMax[v]);
      rowSum[v] *= corr;
#pragma unroll
      for (int t = 0; t < 4; ++t) accO[t][v] *= corr;
      rowMax[v] = newMax[v];
    }

    // ---- P = exp(S*scale - rowMax); stage bf16 P through LDS ----
#pragma unroll
    for (int nt = 0; nt < 4; ++nt)
#pragma unroll
      for (int v = 0; v < 8; ++v) {
        const float p = __expf(sAcc[nt][v] * scale - rowMax[v]);
        rowSum[v] += p;  // per-lane partial; reduced at the end
        pLDS[wave][halfSel * 8 + v][nt * 16 + rIT] = (__bf16)p;
      }

    asm volatile("s_wait_dscnt 0" ::: "memory");

    Frag pa[2];
#pragma unroll
    for (int c = 0; c < 2; ++c) {
      const __bf16* lp = &pLDS[wave][rIT][c * 32 + halfSel * 8];
      pa[c].u[0] = *reinterpret_cast<const uint4*>(lp);
      pa[c].u[1] = *reinterpret_cast<const uint4*>(lp + 16);
    }

    // ---- accO += P * V, V fragments from the LDS tile ([d][key]) ----
#pragma unroll
    for (int nt2 = 0; nt2 < 4; ++nt2) {
      Frag vb0, vb1;
      load_b_frag_lds(vb0, &vTile[buf][0][0], nt2 * 16 + rIT, 0, halfSel);
      load_b_frag_lds(vb1, &vTile[buf][0][0], nt2 * 16 + rIT, 32, halfSel);
      accO[nt2] = wmma_bf16(pa[0], vb0, accO[nt2]);
      accO[nt2] = wmma_bf16(pa[1], vb1, accO[nt2]);
    }

    __syncthreads();  // all waves done reading before TDM overwrites this buffer
  }

  // ---- finalize: reduce row sums across the 16-lane group, normalize, store ----
  float inv[8];
#pragma unroll
  for (int v = 0; v < 8; ++v) {
    float s = rowSum[v];
#pragma unroll
    for (int off = 1; off < 16; off <<= 1) s += __shfl_xor(s, off, 32);
    inv[v] = 1.0f / s;
  }

  const int b = bh >> 4, h = bh & (HH - 1);
#pragma unroll
  for (int nt2 = 0; nt2 < 4; ++nt2)
#pragma unroll
    for (int v = 0; v < 8; ++v) {
      const int s = qRowBase + halfSel * 8 + v;
      const int col = h * DK + nt2 * 16 + rIT;
      O[(size_t)(b * SS + s) * DD + col] = (__bf16)(accO[nt2][v] * inv[v]);
    }
}

// ---------------------------------------------------------------------------
// Host-side orchestration
// ---------------------------------------------------------------------------
extern "C" void kernel_launch(void* const* d_in, const int* in_sizes, int n_in,
                              void* d_out, int out_size, void* d_ws, size_t ws_size,
                              hipStream_t stream) {
  const float* x  = (const float*)d_in[0];
  const float* Wq = (const float*)d_in[1];
  const float* Wk = (const float*)d_in[2];
  const float* Wv = (const float*)d_in[3];
  const float* Wo = (const float*)d_in[4];

  __bf16* ws = (__bf16*)d_ws;
  const size_t nX = (size_t)MM * DD;   // 4M
  const size_t nW = (size_t)DD * DD;   // 1M
  __bf16* xb  = ws;
  __bf16* wqb = xb  + nX;
  __bf16* wkb = wqb + nW;
  __bf16* wvb = wkb + nW;
  __bf16* wob = wvb + nW;
  __bf16* qb  = wob + nW;
  __bf16* kb  = qb  + nX;
  __bf16* vtb = kb  + nX;
  __bf16* ab  = xb;                    // reuse x's bf16 region for attention out

  // 1) conversions
  cvt_f32_bf16<<<(int)((nX + 255) / 256), 256, 0, stream>>>(x,  xb,  (int)nX);
  cvt_f32_bf16<<<(int)((nW + 255) / 256), 256, 0, stream>>>(Wq, wqb, (int)nW);
  cvt_f32_bf16<<<(int)((nW + 255) / 256), 256, 0, stream>>>(Wk, wkb, (int)nW);
  cvt_f32_bf16<<<(int)((nW + 255) / 256), 256, 0, stream>>>(Wv, wvb, (int)nW);
  cvt_f32_bf16<<<(int)((nW + 255) / 256), 256, 0, stream>>>(Wo, wob, (int)nW);

  // 2) QKV projections (WMMA GEMMs)
  dim3 gGemm(MM / 64, DD / 64);
  gemm_bf16<<<gGemm, 128, 0, stream>>>(xb, wqb, qb,  MM, DD, DD, 0);
  gemm_bf16<<<gGemm, 128, 0, stream>>>(xb, wkb, kb,  MM, DD, DD, 0);
  gemm_bf16<<<gGemm, 128, 0, stream>>>(xb, wvb, vtb, MM, DD, DD, 1);

  // 3) flash attention core (TDM-staged K/V tiles)
  dim3 gAttn(BB * HH, SS / 64);
  flash_attn<<<gAttn, 128, 0, stream>>>(qb, kb, vtb, ab);

  // 4) output projection -> f32 d_out
  gemm_bf16<<<gGemm, 128, 0, stream>>>(ab, wob, d_out, MM, DD, DD, 2);
}
